// GATfly3_50285477101906
// MI455X (gfx1250) — matrix operand order repaired and checked by reading, not compiled, not run on hardware
//
#include <hip/hip_runtime.h>
#include <hip/hip_bf16.h>
#include <math.h>

// ---------------------------------------------------------------------------
// Types for CDNA5 WMMA (wave32): v_wmma_f32_16x16x32_bf16
// ---------------------------------------------------------------------------
typedef __attribute__((ext_vector_type(16))) __bf16 bf16x16;
typedef __attribute__((ext_vector_type(8)))  float  f32x8;

__device__ __forceinline__ float gelu_f(float x) {
    return 0.5f * x * (1.0f + erff(x * 0.7071067811865476f));
}

__device__ __forceinline__ unsigned f2ord(float f) {
    unsigned u = __float_as_uint(f);
    return (u & 0x80000000u) ? ~u : (u | 0x80000000u);
}
__device__ __forceinline__ float ord2f(unsigned u) {
    return __uint_as_float((u & 0x80000000u) ? (u & 0x7FFFFFFFu) : ~u);
}

__device__ __forceinline__ float wave_sum(float v) {
    for (int off = 16; off; off >>= 1) v += __shfl_down(v, off, 32);
    return v;
}

// ---------------------------------------------------------------------------
// Operand packing: fp32 -> split bf16 (hi + lo) in WMMA-fragment-native order.
// Fragment layouts per CDNA5 ISA 7.12.2 (wave32):
//   A 16x32: lane half selects K-phase; elem i -> K = i + (i>=8)*8 + (lane>=16)*8
//   B 32x16: elem i -> K = i + (i>=8)*8 + (lane>=16)*8
// Packed element index: ((tile * KC + kc) * 32 + lane) * 16 + i
// K tail is zero-padded here so the GEMM loop is branch-free.
// ---------------------------------------------------------------------------
__global__ void k_pack_a(const float* __restrict__ A, __bf16* __restrict__ ahi,
                         __bf16* __restrict__ alo, int M, int K, int KC)
{
    long long idx   = (long long)blockIdx.x * blockDim.x + threadIdx.x;
    long long total = (long long)(M >> 4) * KC * 512;
    if (idx >= total) return;
    int i    = (int)(idx & 15);
    int lane = (int)((idx >> 4) & 31);
    int kc   = (int)((idx >> 9) % KC);
    int tm   = (int)(idx / ((long long)KC << 9));
    int half = lane >> 4, sub = lane & 15;
    int ka   = (kc << 5) + i + ((i >= 8) ? 8 : 0) + (half << 3);
    int m    = (tm << 4) + sub;
    float v  = (ka < K) ? A[(size_t)m * K + ka] : 0.f;
    __bf16 h = (__bf16)v;
    ahi[idx] = h;
    alo[idx] = (__bf16)(v - (float)h);
}

__global__ void k_pack_b(const float* __restrict__ W, __bf16* __restrict__ bhi,
                         __bf16* __restrict__ blo, int K, int N, int KC)
{
    long long idx   = (long long)blockIdx.x * blockDim.x + threadIdx.x;
    long long total = (long long)(N >> 4) * KC * 512;
    if (idx >= total) return;
    int i    = (int)(idx & 15);
    int lane = (int)((idx >> 4) & 31);
    int kc   = (int)((idx >> 9) % KC);
    int tn   = (int)(idx / ((long long)KC << 9));
    int half = lane >> 4, sub = lane & 15;
    int kb   = (kc << 5) + i + ((i >= 8) ? 8 : 0) + (half << 3);
    int col  = (tn << 4) + sub;
    float v  = (kb < K) ? W[(size_t)kb * N + col] : 0.f;
    __bf16 h = (__bf16)v;
    bhi[idx] = h;
    blo[idx] = (__bf16)(v - (float)h);
}

// ---------------------------------------------------------------------------
// Packed split-precision bf16 WMMA GEMM: C[M,N] = A @ W (+bias, +GELU)
// acc += a_hi*b_hi + a_hi*b_lo + a_lo*b_hi  (~2^-16 rel error, fp32 acc)
// One wave computes a 16x64 strip (1 M-tile x 4 N-tiles): the A fragment is
// loaded once per k-chunk and reused by 4 B fragments -> 12 WMMA per chunk.
// All fragment loads are contiguous b128s; no conversion VALU in the loop.
// blockDim.x = 256 (8 waves). Requires (N/16) % 4 == 0, M % 16 == 0.
// ---------------------------------------------------------------------------
__global__ void k_gemm_wmma_packed(const __bf16* __restrict__ ahi, const __bf16* __restrict__ alo,
                                   const __bf16* __restrict__ bhi, const __bf16* __restrict__ blo,
                                   const float* __restrict__ bias, float* __restrict__ C,
                                   int M, int N, int KC, int act)
{
    const int lane   = threadIdx.x & 31;
    const int wave   = blockIdx.x * (blockDim.x >> 5) + (threadIdx.x >> 5);
    const int tilesN = N >> 4;
    const int tilesM = M >> 4;
    const int ngrp   = tilesN >> 2;
    if (wave >= tilesM * ngrp) return;            // wave-uniform: EXEC all-1s at WMMA
    const int tm = wave / ngrp;
    const int tg = wave % ngrp;

    const size_t astride = 512;                   // elements per (tile,kc) block
    const size_t abase   = ((size_t)tm * KC * 32 + lane) * 16;
    const size_t bbase0  = (((size_t)(tg * 4 + 0) * KC) * 32 + lane) * 16;
    const size_t bbase1  = (((size_t)(tg * 4 + 1) * KC) * 32 + lane) * 16;
    const size_t bbase2  = (((size_t)(tg * 4 + 2) * KC) * 32 + lane) * 16;
    const size_t bbase3  = (((size_t)(tg * 4 + 3) * KC) * 32 + lane) * 16;

    f32x8 acc0 = {0.f,0.f,0.f,0.f,0.f,0.f,0.f,0.f};
    f32x8 acc1 = acc0, acc2 = acc0, acc3 = acc0;

    for (int kc = 0; kc < KC; ++kc) {
        const size_t ko = (size_t)kc * astride;
        if (kc + 1 < KC) {                         // prefetch next A chunk (global_prefetch_b8)
            __builtin_prefetch(ahi + abase + ko + astride, 0, 3);
            __builtin_prefetch(alo + abase + ko + astride, 0, 3);
        }
        const bf16x16 a_hi = *(const bf16x16*)(ahi + abase + ko);
        const bf16x16 a_lo = *(const bf16x16*)(alo + abase + ko);

        {
            const bf16x16 b_hi = *(const bf16x16*)(bhi + bbase0 + ko);
            const bf16x16 b_lo = *(const bf16x16*)(blo + bbase0 + ko);
            acc0 = __builtin_amdgcn_wmma_f32_16x16x32_bf16(false, a_hi, false, b_hi, (short)0, acc0, false, false);
            acc0 = __builtin_amdgcn_wmma_f32_16x16x32_bf16(false, a_hi, false, b_lo, (short)0, acc0, false, false);
            acc0 = __builtin_amdgcn_wmma_f32_16x16x32_bf16(false, a_lo, false, b_hi, (short)0, acc0, false, false);
        }
        {
            const bf16x16 b_hi = *(const bf16x16*)(bhi + bbase1 + ko);
            const bf16x16 b_lo = *(const bf16x16*)(blo + bbase1 + ko);
            acc1 = __builtin_amdgcn_wmma_f32_16x16x32_bf16(false, a_hi, false, b_hi, (short)0, acc1, false, false);
            acc1 = __builtin_amdgcn_wmma_f32_16x16x32_bf16(false, a_hi, false, b_lo, (short)0, acc1, false, false);
            acc1 = __builtin_amdgcn_wmma_f32_16x16x32_bf16(false, a_lo, false, b_hi, (short)0, acc1, false, false);
        }
        {
            const bf16x16 b_hi = *(const bf16x16*)(bhi + bbase2 + ko);
            const bf16x16 b_lo = *(const bf16x16*)(blo + bbase2 + ko);
            acc2 = __builtin_amdgcn_wmma_f32_16x16x32_bf16(false, a_hi, false, b_hi, (short)0, acc2, false, false);
            acc2 = __builtin_amdgcn_wmma_f32_16x16x32_bf16(false, a_hi, false, b_lo, (short)0, acc2, false, false);
            acc2 = __builtin_amdgcn_wmma_f32_16x16x32_bf16(false, a_lo, false, b_hi, (short)0, acc2, false, false);
        }
        {
            const bf16x16 b_hi = *(const bf16x16*)(bhi + bbase3 + ko);
            const bf16x16 b_lo = *(const bf16x16*)(blo + bbase3 + ko);
            acc3 = __builtin_amdgcn_wmma_f32_16x16x32_bf16(false, a_hi, false, b_hi, (short)0, acc3, false, false);
            acc3 = __builtin_amdgcn_wmma_f32_16x16x32_bf16(false, a_hi, false, b_lo, (short)0, acc3, false, false);
            acc3 = __builtin_amdgcn_wmma_f32_16x16x32_bf16(false, a_lo, false, b_hi, (short)0, acc3, false, false);
        }
    }

    // C layout: VGPR v, lane L -> row = v + (L>=16)*8, col = (L&15)
    const int half = lane >> 4, sub = lane & 15;
    const int crow = (tm << 4) + (half << 3);
    f32x8 accs[4] = {acc0, acc1, acc2, acc3};
#pragma unroll
    for (int t = 0; t < 4; ++t) {
        const int ccol = ((tg * 4 + t) << 4) + sub;
        const float bv = bias ? bias[ccol] : 0.f;
#pragma unroll
        for (int v = 0; v < 8; ++v) {
            float val = accs[t][v] + bv;
            if (act) val = gelu_f(val);
            C[(size_t)(crow + v) * N + ccol] = val;
        }
    }
}

// ---------------------------------------------------------------------------
// Elementwise / graph kernels
// ---------------------------------------------------------------------------
__global__ void k_fill_f(float* p, float v, int n) {
    int i = blockIdx.x * blockDim.x + threadIdx.x;
    if (i < n) p[i] = v;
}
__global__ void k_fill_u(unsigned* p, unsigned v, int n) {
    int i = blockIdx.x * blockDim.x + threadIdx.x;
    if (i < n) p[i] = v;
}
__global__ void k_count(const int* __restrict__ batch, float* cnt, int N) {
    int n = blockIdx.x * blockDim.x + threadIdx.x;
    if (n < N) atomicAdd(&cnt[batch[n]], 1.f);
}

// per-(node,head): a_s = <h, att_s>, a_d = <h, att_d>; one wave each
__global__ void k_att_scores(const float* __restrict__ h, const float* __restrict__ atts,
                             const float* __restrict__ attd, float* __restrict__ as_,
                             float* __restrict__ ad_, int N, int H, int C)
{
    int wid  = blockIdx.x * (blockDim.x >> 5) + (threadIdx.x >> 5);
    int lane = threadIdx.x & 31;
    if (wid >= N * H) return;
    int n = wid / H, hd = wid % H;
    const float* hp = h + (size_t)n * H * C + (size_t)hd * C;
    const float* sp = atts + (size_t)hd * C;
    const float* dp = attd + (size_t)hd * C;
    float s = 0.f, d = 0.f;
    for (int c = lane; c < C; c += 32) { float hv = hp[c]; s += hv * sp[c]; d += hv * dp[c]; }
    s = wave_sum(s); d = wave_sum(d);
    if (lane == 0) { as_[wid] = s; ad_[wid] = d; }
}

__global__ void k_edge_logits(const int* __restrict__ src, const int* __restrict__ dst,
                              const float* __restrict__ as_, const float* __restrict__ ad_,
                              float* __restrict__ ebuf, unsigned* __restrict__ mbits,
                              int E, int H)
{
    int idx = blockIdx.x * blockDim.x + threadIdx.x;
    if (idx >= E * H) return;
    int e = idx / H, hd = idx % H;
    float v = as_[src[e] * H + hd] + ad_[dst[e] * H + hd];
    v = (v > 0.f) ? v : 0.2f * v;                       // leaky_relu slope 0.2
    ebuf[idx] = v;
    atomicMax(&mbits[dst[e] * H + hd], f2ord(v));       // segment max
}

__global__ void k_edge_exp(const int* __restrict__ dst, float* __restrict__ ebuf,
                           const unsigned* __restrict__ mbits, float* __restrict__ ssum,
                           int E, int H)
{
    int idx = blockIdx.x * blockDim.x + threadIdx.x;
    if (idx >= E * H) return;
    int e = idx / H, hd = idx % H;
    float m  = ord2f(mbits[dst[e] * H + hd]);
    float ex = expf(ebuf[idx] - m);
    ebuf[idx] = ex;
    atomicAdd(&ssum[dst[e] * H + hd], ex);              // segment sum
}

__global__ void k_rowbias(float* __restrict__ out, const float* __restrict__ bias,
                          long long total, int F)
{
    long long idx = (long long)blockIdx.x * blockDim.x + threadIdx.x;
    if (idx < total) out[idx] = bias[(int)(idx % F)];
}

__global__ void k_aggregate(const int* __restrict__ src, const int* __restrict__ dst,
                            const float* __restrict__ ebuf, const float* __restrict__ ssum,
                            const float* __restrict__ h, float* __restrict__ out,
                            int E, int H, int C)
{
    long long idx   = (long long)blockIdx.x * blockDim.x + threadIdx.x;
    long long total = (long long)E * H * C;
    if (idx >= total) return;
    int HC = H * C;
    int f  = (int)(idx % HC);
    int e  = (int)(idx / HC);
    int hd = f / C;
    int d  = dst[e];
    float alpha = ebuf[e * H + hd] / (ssum[d * H + hd] + 1e-16f);
    atomicAdd(&out[(size_t)d * HC + f], alpha * h[(size_t)src[e] * HC + f]);
}

// GraphNorm (PyG 'graph' mode)
__global__ void k_seg_sum(const float* __restrict__ x, const int* __restrict__ batch,
                          float* __restrict__ gsum, int N, int F)
{
    int wid = blockIdx.x * (blockDim.x >> 5) + (threadIdx.x >> 5);
    int lane = threadIdx.x & 31;
    if (wid >= N) return;
    const float* xp = x + (size_t)wid * F;
    float s = 0.f;
    for (int c = lane; c < F; c += 32) s += xp[c];
    s = wave_sum(s);
    if (lane == 0) atomicAdd(&gsum[batch[wid]], s);
}
__global__ void k_seg_var(const float* __restrict__ x, const int* __restrict__ batch,
                          const float* __restrict__ meanb, float* __restrict__ gvar,
                          int N, int F)
{
    int wid = blockIdx.x * (blockDim.x >> 5) + (threadIdx.x >> 5);
    int lane = threadIdx.x & 31;
    if (wid >= N) return;
    const float m   = meanb[batch[wid]];
    const float* xp = x + (size_t)wid * F;
    float s = 0.f;
    for (int c = lane; c < F; c += 32) { float d = xp[c] - m; s += d * d; }
    s = wave_sum(s);
    if (lane == 0) atomicAdd(&gvar[batch[wid]], s);
}
__global__ void k_mean(const float* gsum, const float* cnt, float* meanb, int B, int F) {
    int b = blockIdx.x * blockDim.x + threadIdx.x;
    if (b < B) meanb[b] = gsum[b] / (fmaxf(cnt[b], 1.f) * (float)F);
}
__global__ void k_rstd(const float* gvar, const float* cnt, float* rstdb, int B, int F) {
    int b = blockIdx.x * blockDim.x + threadIdx.x;
    if (b < B) rstdb[b] = rsqrtf(gvar[b] / (fmaxf(cnt[b], 1.f) * (float)F) + 1e-5f);
}
__global__ void k_norm_gelu(float* __restrict__ x, const int* __restrict__ batch,
                            const float* __restrict__ meanb, const float* __restrict__ rstdb,
                            const float* __restrict__ gamma, const float* __restrict__ beta,
                            int N, int F)
{
    long long idx = (long long)blockIdx.x * blockDim.x + threadIdx.x;
    if (idx >= (long long)N * F) return;
    int n = (int)(idx / F), f = (int)(idx % F);
    int b = batch[n];
    float v = (x[idx] - meanb[b]) * rstdb[b];
    x[idx] = gelu_f(v * gamma[f] + beta[f]);
}

// global mean pool
__global__ void k_pool_sum(const float* __restrict__ x, const int* __restrict__ batch,
                           float* __restrict__ psum, int N, int F)
{
    long long idx = (long long)blockIdx.x * blockDim.x + threadIdx.x;
    if (idx >= (long long)N * F) return;
    int n = (int)(idx / F), f = (int)(idx % F);
    atomicAdd(&psum[(size_t)batch[n] * F + f], x[idx]);
}
__global__ void k_pool_div(const float* psum, const float* cnt, float* pool, int B, int F) {
    int idx = blockIdx.x * blockDim.x + threadIdx.x;
    if (idx >= B * F) return;
    pool[idx] = psum[idx] / fmaxf(cnt[idx / F], 1.f);
}

// ---------------------------------------------------------------------------
// Host launch
// ---------------------------------------------------------------------------
extern "C" void kernel_launch(void* const* d_in, const int* in_sizes, int n_in,
                              void* d_out, int out_size, void* d_ws, size_t ws_size,
                              hipStream_t stream)
{
    (void)n_in; (void)out_size; (void)ws_size;
    const float* x    = (const float*)d_in[0];
    const int*   eidx = (const int*)d_in[1];
    const int*   batch= (const int*)d_in[2];
    // d_in[3] = edge_attr, ignored by the reference (edge_dim=None)
    const float* w1 = (const float*)d_in[4];  const float* as1 = (const float*)d_in[5];
    const float* ad1= (const float*)d_in[6];  const float* b1  = (const float*)d_in[7];
    const float* w2 = (const float*)d_in[8];  const float* as2 = (const float*)d_in[9];
    const float* ad2= (const float*)d_in[10]; const float* b2  = (const float*)d_in[11];
    const float* w3 = (const float*)d_in[12]; const float* as3 = (const float*)d_in[13];
    const float* ad3= (const float*)d_in[14]; const float* b3  = (const float*)d_in[15];
    const float* g1 = (const float*)d_in[16]; const float* be1 = (const float*)d_in[17];
    const float* g2 = (const float*)d_in[18]; const float* be2 = (const float*)d_in[19];
    const float* g3 = (const float*)d_in[20]; const float* be3 = (const float*)d_in[21];
    const float* rw = (const float*)d_in[22]; const float* rb  = (const float*)d_in[23];
    const float* l1w= (const float*)d_in[24]; const float* l1b = (const float*)d_in[25];
    const float* l2w= (const float*)d_in[26]; const float* l2b = (const float*)d_in[27];
    const float* ow = (const float*)d_in[28]; const float* ob  = (const float*)d_in[29];

    const int N  = in_sizes[2];
    const int E  = in_sizes[1] / 2;
    const int IN = in_sizes[0] / N;
    const int B  = 16, C = 512;
    const int* src = eidx;          // edge_index[0]
    const int* dst = eidx + E;      // edge_index[1]

    // workspace carve (256B aligned slices); total ~180 MB at N = 8000
    char* wsp = (char*)d_ws;
    auto carve = [&](size_t bytes) { void* p = wsp; wsp += (bytes + 255) & ~(size_t)255; return p; };
    float*    hbuf  = (float*)carve((size_t)N * 1536 * 4);
    float*    obuf  = (float*)carve((size_t)N * 1536 * 4);
    // packed operand planes (sized for the largest use: A = 8000 x KC66*32, B = 96 tiles x KC66)
    const size_t APACK_MAX = (size_t)(8000 / 16) * 66 * 512;   // >= tilesM*KC*512 for all calls
    const size_t BPACK_MAX = (size_t)96 * 66 * 512;            // >= tilesN*KC*512 for all calls
    __bf16*   apackh = (__bf16*)carve(APACK_MAX * 2);
    __bf16*   apackl = (__bf16*)carve(APACK_MAX * 2);
    __bf16*   bpackh = (__bf16*)carve(BPACK_MAX * 2);
    __bf16*   bpackl = (__bf16*)carve(BPACK_MAX * 2);
    float*    asb   = (float*)carve((size_t)N * 3 * 4);
    float*    adb   = (float*)carve((size_t)N * 3 * 4);
    float*    ebuf  = (float*)carve((size_t)E * 3 * 4);
    unsigned* mb    = (unsigned*)carve((size_t)N * 3 * 4);
    float*    ssum  = (float*)carve((size_t)N * 3 * 4);
    float*    cnt   = (float*)carve(B * 4);
    float*    gsum  = (float*)carve(B * 4);
    float*    gvar  = (float*)carve(B * 4);
    float*    meanb = (float*)carve(B * 4);
    float*    rstdb = (float*)carve(B * 4);
    float*    psum  = (float*)carve((size_t)B * 1536 * 4);
    float*    pool  = (float*)carve((size_t)B * 1536 * 4);
    float*    p1    = (float*)carve((size_t)B * 512 * 4);
    float*    p2    = (float*)carve((size_t)B * 256 * 4);
    float*    p3    = (float*)carve((size_t)B * 128 * 4);

    auto cdiv = [](long long a, long long b) { return (int)((a + b - 1) / b); };
    const unsigned ORD_NEG_INF = 0x007FFFFFu;   // f2ord(-inf)

    // pack + GEMM (requires M%16==0 and (N/16)%4==0; true for all calls here)
    auto gemm = [&](const float* A, const float* Wm, const float* bias, float* Cc,
                    int M, int Nc, int K, int act) {
        const int KC = (K + 31) / 32;
        const long long atot = (long long)(M >> 4) * KC * 512;
        const long long btot = (long long)(Nc >> 4) * KC * 512;
        k_pack_a<<<cdiv(atot, 256), 256, 0, stream>>>(A, apackh, apackl, M, K, KC);
        k_pack_b<<<cdiv(btot, 256), 256, 0, stream>>>(Wm, bpackh, bpackl, K, Nc, KC);
        const int waves = (M >> 4) * ((Nc >> 4) >> 2);
        k_gemm_wmma_packed<<<cdiv(waves, 8), 256, 0, stream>>>(apackh, apackl, bpackh, bpackl,
                                                               bias, Cc, M, Nc, KC, act);
    };

    // per-graph node counts (fixed across layers)
    k_fill_f<<<1, 32, 0, stream>>>(cnt, 0.f, B);
    k_count<<<cdiv(N, 256), 256, 0, stream>>>(batch, cnt, N);

    auto gat = [&](const float* X, int Fin, const float* Wm, const float* atts,
                   const float* attd, const float* bias, int H) {
        int HC = H * C;
        gemm(X, Wm, nullptr, hbuf, N, HC, Fin, 0);
        k_att_scores<<<cdiv((long long)N * H, 8), 256, 0, stream>>>(hbuf, atts, attd, asb, adb, N, H, C);
        k_fill_u<<<cdiv((long long)N * H, 256), 256, 0, stream>>>(mb, ORD_NEG_INF, N * H);
        k_fill_f<<<cdiv((long long)N * H, 256), 256, 0, stream>>>(ssum, 0.f, N * H);
        k_edge_logits<<<cdiv((long long)E * H, 256), 256, 0, stream>>>(src, dst, asb, adb, ebuf, mb, E, H);
        k_edge_exp<<<cdiv((long long)E * H, 256), 256, 0, stream>>>(dst, ebuf, mb, ssum, E, H);
        k_rowbias<<<cdiv((long long)N * HC, 256), 256, 0, stream>>>(obuf, bias, (long long)N * HC, HC);
        k_aggregate<<<cdiv((long long)E * HC, 256), 256, 0, stream>>>(src, dst, ebuf, ssum, hbuf, obuf, E, H, C);
    };

    auto gnorm = [&](float* Xio, int F, const float* gamma, const float* beta) {
        k_fill_f<<<1, 32, 0, stream>>>(gsum, 0.f, B);
        k_fill_f<<<1, 32, 0, stream>>>(gvar, 0.f, B);
        k_seg_sum<<<cdiv(N, 8), 256, 0, stream>>>(Xio, batch, gsum, N, F);
        k_mean<<<1, 32, 0, stream>>>(gsum, cnt, meanb, B, F);
        k_seg_var<<<cdiv(N, 8), 256, 0, stream>>>(Xio, batch, meanb, gvar, N, F);
        k_rstd<<<1, 32, 0, stream>>>(gvar, cnt, rstdb, B, F);
        k_norm_gelu<<<cdiv((long long)N * F, 256), 256, 0, stream>>>(Xio, batch, meanb, rstdb, gamma, beta, N, F);
    };

    gat(x,    IN,  w1, as1, ad1, b1, 1);  gnorm(obuf, 512,  g1, be1);
    gat(obuf, 512, w2, as2, ad2, b2, 1);  gnorm(obuf, 512,  g2, be2);
    gat(obuf, 512, w3, as3, ad3, b3, 3);  gnorm(obuf, 1536, g3, be3);

    // global mean pool
    k_fill_f<<<cdiv(B * 1536, 256), 256, 0, stream>>>(psum, 0.f, B * 1536);
    k_pool_sum<<<cdiv((long long)N * 1536, 256), 256, 0, stream>>>(obuf, batch, psum, N, 1536);
    k_pool_div<<<cdiv(B * 1536, 256), 256, 0, stream>>>(psum, cnt, pool, B, 1536);

    // MLP head (all WMMA, M = 16)
    gemm(pool, rw,  rb,  p1,            B, 512, 1536, 1);
    gemm(p1,   l1w, l1b, p2,            B, 256, 512,  1);
    gemm(p2,   l2w, l2b, p3,            B, 128, 256,  1);
    gemm(p3,   ow,  ob,  (float*)d_out, B, 256, 128,  0);
}
